// REHGL_53403623358977
// MI455X (gfx1250) — matrix operand order repaired and checked by reading, not compiled, not run on hardware
//
#include <hip/hip_runtime.h>
#include <hip/hip_bf16.h>
#include <stdint.h>
#include <stddef.h>

#define NT   4096
#define NA_  4096
#define NS_  1024
#define NALL 9216
#define FEAT 128
#define COM  64
#define MPD  64
#define EMB  64
#define NCLS 3

typedef __attribute__((ext_vector_type(16))) __bf16 v16bf;
typedef __attribute__((ext_vector_type(8)))  __bf16 v8bf;
typedef __attribute__((ext_vector_type(8)))  float  v8f;

// ---------------------------------------------------------------------------
// WMMA fragment loaders (layouts per CDNA5 ISA 7.12.2, wave32)
// A: 16x32 bf16, row-major source [*, lda]; lane&15 = M row,
//    lane<16 -> K in {0..7, 16..23}; lane>=16 -> K in {8..15, 24..31}
// B: 32x16 bf16 supplied as Bt[N, K] row-major; lane&15 = N col,
//    lane<16 -> K in {0..7, 16..23}; lane>=16 -> K in {8..15, 24..31}
// ---------------------------------------------------------------------------
__device__ __forceinline__ v16bf ld_a_frag(const __bf16* A, int lda, int mBase,
                                           int kk, int lane) {
  const int m  = mBase + (lane & 15);
  const int bk = (lane < 16) ? 0 : 8;
  const __bf16* p = A + (size_t)m * lda + kk + bk;
  v8bf lo = *(const v8bf*)(p);
  v8bf hi = *(const v8bf*)(p + 16);
  v16bf a;
#pragma unroll
  for (int j = 0; j < 8; ++j) { a[j] = lo[j]; a[8 + j] = hi[j]; }
  return a;
}

__device__ __forceinline__ v16bf ld_b_frag(const __bf16* Bt, int ldb, int nBase,
                                           int kk, int lane) {
  const int n  = nBase + (lane & 15);
  const int bk = (lane < 16) ? 0 : 8;
  const __bf16* p = Bt + (size_t)n * ldb + kk + bk;
  v8bf lo = *(const v8bf*)(p);
  v8bf hi = *(const v8bf*)(p + 16);
  v16bf b;
#pragma unroll
  for (int j = 0; j < 8; ++j) { b[j] = lo[j]; b[8 + j] = hi[j]; }
  return b;
}

// ---------------------------------------------------------------------------
// C = threshold( (Y Y^T) * invH, th ), Y bf16 [N, K].
// Register-blocked: each wave computes a 32x32 tile (2x2 WMMA frags, 4 accs),
// 4 waves/block in a 2x2 arrangement -> 64x64 per block.
// Requires N % 64 == 0, K % 32 == 0.
// ---------------------------------------------------------------------------
__global__ void simgemm_kernel(const __bf16* __restrict__ Y, int N, int K,
                               float invH, float th,
                               float* __restrict__ Cf, __bf16* __restrict__ Cb) {
  const int lane  = threadIdx.x & 31;
  const int wid   = threadIdx.x >> 5;
  const int mBase = blockIdx.y * 64 + (wid >> 1) * 32;
  const int nBase = blockIdx.x * 64 + (wid & 1) * 32;
  if (nBase >= N || mBase >= N) return;
  v8f c00 = {}, c01 = {}, c10 = {}, c11 = {};
  for (int kk = 0; kk < K; kk += 32) {
    v16bf a0 = ld_a_frag(Y, K, mBase,      kk, lane);
    v16bf a1 = ld_a_frag(Y, K, mBase + 16, kk, lane);
    v16bf b0 = ld_b_frag(Y, K, nBase,      kk, lane);
    v16bf b1 = ld_b_frag(Y, K, nBase + 16, kk, lane);
    c00 = __builtin_amdgcn_wmma_f32_16x16x32_bf16(false, a0, false, b0, (short)0, c00, false, false);
    c01 = __builtin_amdgcn_wmma_f32_16x16x32_bf16(false, a0, false, b1, (short)0, c01, false, false);
    c10 = __builtin_amdgcn_wmma_f32_16x16x32_bf16(false, a1, false, b0, (short)0, c10, false, false);
    c11 = __builtin_amdgcn_wmma_f32_16x16x32_bf16(false, a1, false, b1, (short)0, c11, false, false);
  }
  const int nn   = lane & 15;
  const int mOff = (lane < 16) ? 0 : 8;
#pragma unroll
  for (int mt = 0; mt < 2; ++mt) {
#pragma unroll
    for (int nt = 0; nt < 2; ++nt) {
      const v8f& c = mt ? (nt ? c11 : c10) : (nt ? c01 : c00);
      const int n = nBase + nt * 16 + nn;
      const int m = mBase + mt * 16 + mOff;
#pragma unroll
      for (int r = 0; r < 8; ++r) {
        float s = c[r] * invH;
        s = (s < th) ? 0.0f : s;
        size_t idx = (size_t)(m + r) * (size_t)N + n;
        if (Cf) Cf[idx] = s;
        if (Cb) Cb[idx] = (__bf16)s;
      }
    }
  }
}

// ---------------------------------------------------------------------------
// Generic C[M,N] f32 = A[M,K] bf16 @ Bt[N,K]^T bf16 (+bias)(+relu)
// Same 32x32-per-wave / 64x64-per-block register blocking.
// Requires M % 64 == 0, N % 64 == 0, K % 32 == 0.
// ---------------------------------------------------------------------------
__global__ void gemm_kernel(const __bf16* __restrict__ A,
                            const __bf16* __restrict__ Bt,
                            int M, int N, int K,
                            const float* __restrict__ bias, int relu,
                            float* __restrict__ C) {
  const int lane  = threadIdx.x & 31;
  const int wid   = threadIdx.x >> 5;
  const int mBase = blockIdx.y * 64 + (wid >> 1) * 32;
  const int nBase = blockIdx.x * 64 + (wid & 1) * 32;
  if (nBase >= N || mBase >= M) return;
  v8f c00 = {}, c01 = {}, c10 = {}, c11 = {};
  for (int kk = 0; kk < K; kk += 32) {
    v16bf a0 = ld_a_frag(A, K, mBase,      kk, lane);
    v16bf a1 = ld_a_frag(A, K, mBase + 16, kk, lane);
    v16bf b0 = ld_b_frag(Bt, K, nBase,      kk, lane);
    v16bf b1 = ld_b_frag(Bt, K, nBase + 16, kk, lane);
    c00 = __builtin_amdgcn_wmma_f32_16x16x32_bf16(false, a0, false, b0, (short)0, c00, false, false);
    c01 = __builtin_amdgcn_wmma_f32_16x16x32_bf16(false, a0, false, b1, (short)0, c01, false, false);
    c10 = __builtin_amdgcn_wmma_f32_16x16x32_bf16(false, a1, false, b0, (short)0, c10, false, false);
    c11 = __builtin_amdgcn_wmma_f32_16x16x32_bf16(false, a1, false, b1, (short)0, c11, false, false);
  }
  const int nn   = lane & 15;
  const int mOff = (lane < 16) ? 0 : 8;
#pragma unroll
  for (int mt = 0; mt < 2; ++mt) {
#pragma unroll
    for (int nt = 0; nt < 2; ++nt) {
      const v8f& c = mt ? (nt ? c11 : c10) : (nt ? c01 : c00);
      const int n = nBase + nt * 16 + nn;
      const int m = mBase + mt * 16 + mOff;
      const float bv = bias ? bias[n] : 0.0f;
#pragma unroll
      for (int r = 0; r < 8; ++r) {
        float s = c[r] + bv;
        if (relu) s = fmaxf(s, 0.0f);
        C[(size_t)(m + r) * (size_t)N + n] = s;
      }
    }
  }
}

// ---------------------------------------------------------------------------
// Row-wise per-head weight + L2 normalize: X f32 [n,d], w f32 [2,d]
// -> Y bf16 [n, 2d]. blockDim = d (64 or 128), grid = n.
// ---------------------------------------------------------------------------
__global__ void rownorm_kernel(const float* __restrict__ X, int d,
                               const float* __restrict__ w,
                               __bf16* __restrict__ Y) {
  __shared__ float r0[128];
  __shared__ float r1[128];
  const int row = blockIdx.x, t = threadIdx.x;
  const float x  = X[(size_t)row * d + t];
  const float v0 = x * w[t];
  const float v1 = x * w[d + t];
  r0[t] = v0 * v0;
  r1[t] = v1 * v1;
  __syncthreads();
  for (int s = d >> 1; s > 0; s >>= 1) {
    if (t < s) { r0[t] += r0[t + s]; r1[t] += r1[t + s]; }
    __syncthreads();
  }
  const float n0 = fmaxf(sqrtf(r0[0]), 1e-12f);
  const float n1 = fmaxf(sqrtf(r1[0]), 1e-12f);
  Y[(size_t)row * (2 * d) + t]     = (__bf16)(v0 / n0);
  Y[(size_t)row * (2 * d) + d + t] = (__bf16)(v1 / n1);
}

// out[c*R + r] = bf16(in[r*C + c])
__global__ void transpose_bf16_kernel(const float* __restrict__ in, int R, int C,
                                      __bf16* __restrict__ out) {
  size_t idx = (size_t)blockIdx.x * blockDim.x + threadIdx.x;
  if (idx >= (size_t)R * C) return;
  int r = (int)(idx / C), c = (int)(idx % C);
  out[(size_t)c * R + r] = (__bf16)in[idx];
}

// out[r*C+c] = bf16(in[r*ld + c])  (strided slice -> packed bf16)
__global__ void conv_bf16_kernel(const float* __restrict__ in, int R, int C,
                                 int ld, __bf16* __restrict__ out) {
  size_t idx = (size_t)blockIdx.x * blockDim.x + threadIdx.x;
  if (idx >= (size_t)R * C) return;
  int r = (int)(idx / C), c = (int)(idx % C);
  out[idx] = (__bf16)in[(size_t)r * ld + c];
}

// cs[j] = sum_i |A[i,j]|
__global__ void colsum_abs_kernel(const float* __restrict__ A, int R, int C,
                                  float* __restrict__ cs) {
  int j = blockIdx.x * blockDim.x + threadIdx.x;
  if (j >= C) return;
  float s = 0.0f;
  for (int i = 0; i < R; ++i) s += fabsf(A[(size_t)i * C + j]);
  cs[j] = s;
}

// dst = (accum ? dst : 0) + softmax(wvec)[widx] * src[i,j] / max(cs[j],1e-12)
__global__ void combine_kernel(const float* __restrict__ src,
                               const float* __restrict__ cs, int R, int C,
                               const float* __restrict__ wvec, int wn, int widx,
                               int accum, float* __restrict__ dst) {
  size_t idx = (size_t)blockIdx.x * blockDim.x + threadIdx.x;
  if (idx >= (size_t)R * C) return;
  float m = wvec[0];
  for (int i = 1; i < wn; ++i) m = fmaxf(m, wvec[i]);
  float den = 0.0f;
  for (int i = 0; i < wn; ++i) den += expf(wvec[i] - m);
  const float wgt = expf(wvec[widx] - m) / den;
  const int j = (int)(idx % C);
  const float v = wgt * src[idx] / fmaxf(cs[j], 1e-12f);
  dst[idx] = accum ? (dst[idx] + v) : v;
}

// O[i,j] = A[i,j] + A[j,i]
__global__ void symmetrize_kernel(const float* __restrict__ A, int N,
                                  float* __restrict__ O) {
  int j = blockIdx.x * blockDim.x + threadIdx.x;
  int i = blockIdx.y;
  if (j >= N) return;
  O[(size_t)i * N + j] = A[(size_t)i * N + j] + A[(size_t)j * N + i];
}

// adj = S[i,j]/max(cs[j],1e-12)  (f32 to d_out + bf16 copy for GCN GEMM)
__global__ void normadj_kernel(const float* __restrict__ S,
                               const float* __restrict__ cs, int N,
                               float* __restrict__ adj_out,
                               __bf16* __restrict__ adj_bf) {
  size_t idx = (size_t)blockIdx.x * blockDim.x + threadIdx.x;
  if (idx >= (size_t)N * N) return;
  const int j = (int)(idx % N);
  const float v = S[idx] / fmaxf(cs[j], 1e-12f);
  adj_out[idx] = v;
  adj_bf[idx]  = (__bf16)v;
}

// O[M,N] = H[M,K] @ W[K,N]  (tiny N, plain VALU)
__global__ void small_gemm_kernel(const float* __restrict__ H,
                                  const float* __restrict__ W,
                                  int M, int K, int N, float* __restrict__ O) {
  int idx = blockIdx.x * blockDim.x + threadIdx.x;
  if (idx >= M * N) return;
  int i = idx / N, c = idx % N;
  float s = 0.0f;
  for (int k = 0; k < K; ++k) s += H[(size_t)i * K + k] * W[(size_t)k * N + c];
  O[idx] = s;
}

// logits[i,:] = log_softmax( adj[i,:] @ X2 + b2 ), NCLS==3. block=128, grid=NT
__global__ void final_logits_kernel(const float* __restrict__ adj,
                                    const float* __restrict__ X2,
                                    const float* __restrict__ b2, int N,
                                    float* __restrict__ logits) {
  __shared__ float fr[3][128];
  const int i = blockIdx.x, t = threadIdx.x;
  float a0 = 0.f, a1 = 0.f, a2 = 0.f;
  for (int k = t; k < N; k += 128) {
    const float av = adj[(size_t)i * N + k];
    a0 += av * X2[k * 3 + 0];
    a1 += av * X2[k * 3 + 1];
    a2 += av * X2[k * 3 + 2];
  }
  fr[0][t] = a0; fr[1][t] = a1; fr[2][t] = a2;
  __syncthreads();
  for (int s = 64; s > 0; s >>= 1) {
    if (t < s) { fr[0][t] += fr[0][t+s]; fr[1][t] += fr[1][t+s]; fr[2][t] += fr[2][t+s]; }
    __syncthreads();
  }
  if (t == 0) {
    float y0 = fr[0][0] + b2[0], y1 = fr[1][0] + b2[1], y2 = fr[2][0] + b2[2];
    float m = fmaxf(y0, fmaxf(y1, y2));
    float lse = m + logf(expf(y0 - m) + expf(y1 - m) + expf(y2 - m));
    logits[i * 3 + 0] = y0 - lse;
    logits[i * 3 + 1] = y1 - lse;
    logits[i * 3 + 2] = y2 - lse;
  }
}

// ---------------------------------------------------------------------------
// Host-side orchestration
// ---------------------------------------------------------------------------
static inline dim3 grid1(size_t total) { return dim3((unsigned)((total + 255) / 256)); }

extern "C" void kernel_launch(void* const* d_in, const int* in_sizes, int n_in,
                              void* d_out, int out_size, void* d_ws, size_t ws_size,
                              hipStream_t stream) {
  (void)in_sizes; (void)n_in; (void)out_size; (void)ws_size;
  const float* features  = (const float*)d_in[0];
  const float* adj_ori   = (const float*)d_in[1];
  const float* mp_pap    = (const float*)d_in[2];
  const float* mp_psp    = (const float*)d_in[3];
  const float* fgo_w     = (const float*)d_in[6];
  const float* fpo_w     = (const float*)d_in[7];
  const float* sgg_pap_w = (const float*)d_in[8];
  const float* sgg_psp_w = (const float*)d_in[9];
  const float* sg_agg_w  = (const float*)d_in[10];
  const float* f_agg_f_w = (const float*)d_in[11];
  const float* f_agg_w   = (const float*)d_in[12];
  const float* topo_W_a  = (const float*)d_in[13];
  const float* topo_b_a  = (const float*)d_in[14];
  const float* topo_W_s  = (const float*)d_in[15];
  const float* topo_b_s  = (const float*)d_in[16];
  const float* fgt_w_a   = (const float*)d_in[17];
  const float* fgt_w_s   = (const float*)d_in[18];
  const float* gcn_W1    = (const float*)d_in[19];
  const float* gcn_b1    = (const float*)d_in[20];
  const float* gcn_W2    = (const float*)d_in[21];
  const float* gcn_b2    = (const float*)d_in[22];

  float* logits  = (float*)d_out;
  float* adj_out = logits + (size_t)NT * NCLS;

  // bump allocator over d_ws
  char* wsp = (char*)d_ws;
  auto alloc = [&wsp](size_t bytes) -> char* {
    char* p = (char*)(((uintptr_t)wsp + 255) & ~(uintptr_t)255);
    wsp = p + bytes;
    return p;
  };
  const size_t N2 = (size_t)NT * NT;
  float*  F_ACC  = (float*)alloc(N2 * 4);
  float*  F_TMP  = (float*)alloc(N2 * 4);
  float*  F_TMP2 = (float*)alloc(N2 * 4);
  float*  CS     = (float*)alloc((size_t)NT * 4);
  __bf16* B_SIMA = (__bf16*)alloc(N2 * 2);                       // sim_a, later new_adj bf16
  __bf16* B_SIMS = (__bf16*)alloc((size_t)NS_ * NS_ * 2);
  __bf16* B_ORIA = (__bf16*)alloc((size_t)NT * NA_ * 2);
  __bf16* B_ORIS = (__bf16*)alloc((size_t)NT * NS_ * 2);
  __bf16* B_Y    = (__bf16*)alloc((size_t)NT * 256 * 2);         // normalized features
  __bf16* B_WT   = (__bf16*)alloc((size_t)COM * NA_ * 2);        // topo_W^T (a then s)
  __bf16* B_FAT  = (__bf16*)alloc((size_t)FEAT * NA_ * 2);       // fmat_a^T
  __bf16* B_FST  = (__bf16*)alloc((size_t)FEAT * NS_ * 2);       // fmat_s^T
  __bf16* B_T1T  = (__bf16*)alloc((size_t)COM * NA_ * 2);        // t1^T
  __bf16* B_FT   = (__bf16*)alloc((size_t)NT * FEAT * 2);        // fmat_targ bf16
  __bf16* B_W1T  = (__bf16*)alloc((size_t)EMB * FEAT * 2);       // gcn_W1^T
  __bf16* B_H0T  = (__bf16*)alloc((size_t)EMB * NT * 2);         // (ft@W1)^T
  float*  F_T1   = (float*)alloc((size_t)NA_ * COM * 4);
  float*  F_THA  = (float*)alloc((size_t)NT * COM * 4);
  float*  F_THS  = (float*)alloc((size_t)NT * COM * 4);
  float*  F_FPA  = (float*)alloc((size_t)NT * FEAT * 4);
  float*  F_FPS  = (float*)alloc((size_t)NT * FEAT * 4);
  float*  F_H0   = (float*)alloc((size_t)NT * EMB * 4);
  float*  F_H1   = (float*)alloc((size_t)NT * EMB * 4);
  float*  F_X2   = (float*)alloc((size_t)NT * 4 * 4);

  // ---- prep: bf16 conversions / transposes ----
  conv_bf16_kernel<<<grid1((size_t)NT*NA_),256,0,stream>>>(adj_ori + NT, NT, NA_, NALL, B_ORIA);
  conv_bf16_kernel<<<grid1((size_t)NT*NS_),256,0,stream>>>(adj_ori + NT + NA_, NT, NS_, NALL, B_ORIS);
  conv_bf16_kernel<<<grid1((size_t)NT*FEAT),256,0,stream>>>(features, NT, FEAT, FEAT, B_FT);
  transpose_bf16_kernel<<<grid1((size_t)NA_*FEAT),256,0,stream>>>(features + (size_t)NT*FEAT, NA_, FEAT, B_FAT);
  transpose_bf16_kernel<<<grid1((size_t)NS_*FEAT),256,0,stream>>>(features + (size_t)(NT+NA_)*FEAT, NS_, FEAT, B_FST);
  transpose_bf16_kernel<<<grid1((size_t)FEAT*EMB),256,0,stream>>>(gcn_W1, FEAT, EMB, B_W1T);

  // ---- relation a: topo_hid_a = ori_a @ (sim_a @ W_a) + b_a ; feat_prop_a ----
  transpose_bf16_kernel<<<grid1((size_t)NA_*COM),256,0,stream>>>(topo_W_a, NA_, COM, B_WT);
  rownorm_kernel<<<NA_,FEAT,0,stream>>>(features + (size_t)NT*FEAT, FEAT, fgo_w, B_Y);
  simgemm_kernel<<<dim3(NA_/64, NA_/64),128,0,stream>>>(B_Y, NA_, 2*FEAT, 0.5f, 0.1f, nullptr, B_SIMA);
  gemm_kernel<<<dim3(1, NA_/64),128,0,stream>>>(B_SIMA, B_WT, NA_, COM, NA_, nullptr, 0, F_T1);
  transpose_bf16_kernel<<<grid1((size_t)NA_*COM),256,0,stream>>>(F_T1, NA_, COM, B_T1T);
  gemm_kernel<<<dim3(1, NT/64),128,0,stream>>>(B_ORIA, B_T1T, NT, COM, NA_, topo_b_a, 0, F_THA);
  gemm_kernel<<<dim3(2, NT/64),128,0,stream>>>(B_ORIA, B_FAT, NT, FEAT, NA_, nullptr, 0, F_FPA);

  // ---- relation s ----
  transpose_bf16_kernel<<<grid1((size_t)NS_*COM),256,0,stream>>>(topo_W_s, NS_, COM, B_WT);
  rownorm_kernel<<<NS_,FEAT,0,stream>>>(features + (size_t)(NT+NA_)*FEAT, FEAT, fgo_w, B_Y);
  simgemm_kernel<<<dim3(NS_/64, NS_/64),128,0,stream>>>(B_Y, NS_, 2*FEAT, 0.5f, 0.1f, nullptr, B_SIMS);
  gemm_kernel<<<dim3(1, NS_/64),128,0,stream>>>(B_SIMS, B_WT, NS_, COM, NS_, nullptr, 0, F_T1);
  transpose_bf16_kernel<<<grid1((size_t)NS_*COM),256,0,stream>>>(F_T1, NS_, COM, B_T1T);
  gemm_kernel<<<dim3(1, NT/64),128,0,stream>>>(B_ORIS, B_T1T, NT, COM, NS_, topo_b_s, 0, F_THS);
  gemm_kernel<<<dim3(2, NT/64),128,0,stream>>>(B_ORIS, B_FST, NT, FEAT, NS_, nullptr, 0, F_FPS);

  const size_t NB = (N2 + 255) / 256;
  // ---- channel 0: g_targ ----
  rownorm_kernel<<<NT,FEAT,0,stream>>>(features, FEAT, fgo_w, B_Y);
  simgemm_kernel<<<dim3(NT/64, NT/64),128,0,stream>>>(B_Y, NT, 2*FEAT, 0.5f, 0.1f, F_TMP, nullptr);
  colsum_abs_kernel<<<NT/256,256,0,stream>>>(F_TMP, NT, NT, CS);
  combine_kernel<<<NB,256,0,stream>>>(F_TMP, CS, NT, NT, f_agg_w, 4, 0, 0, F_ACC);

  // ---- channel 1: sem_g = att([pap, psp], sg_agg_w) ----
  rownorm_kernel<<<NT,MPD,0,stream>>>(mp_pap, MPD, sgg_pap_w, B_Y);
  simgemm_kernel<<<dim3(NT/64, NT/64),128,0,stream>>>(B_Y, NT, 2*MPD, 0.5f, 0.1f, F_TMP, nullptr);
  colsum_abs_kernel<<<NT/256,256,0,stream>>>(F_TMP, NT, NT, CS);
  combine_kernel<<<NB,256,0,stream>>>(F_TMP, CS, NT, NT, sg_agg_w, 2, 0, 0, F_TMP2);
  rownorm_kernel<<<NT,MPD,0,stream>>>(mp_psp, MPD, sgg_psp_w, B_Y);
  simgemm_kernel<<<dim3(NT/64, NT/64),128,0,stream>>>(B_Y, NT, 2*MPD, 0.5f, 0.1f, F_TMP, nullptr);
  colsum_abs_kernel<<<NT/256,256,0,stream>>>(F_TMP, NT, NT, CS);
  combine_kernel<<<NB,256,0,stream>>>(F_TMP, CS, NT, NT, sg_agg_w, 2, 1, 1, F_TMP2);
  colsum_abs_kernel<<<NT/256,256,0,stream>>>(F_TMP2, NT, NT, CS);
  combine_kernel<<<NB,256,0,stream>>>(F_TMP2, CS, NT, NT, f_agg_w, 4, 1, 1, F_ACC);

  // ---- channel 2: fp_g = att([gg(feat_prop_a), gg(feat_prop_s)], f_agg_f_w) ----
  rownorm_kernel<<<NT,FEAT,0,stream>>>(F_FPA, FEAT, fpo_w, B_Y);
  simgemm_kernel<<<dim3(NT/64, NT/64),128,0,stream>>>(B_Y, NT, 2*FEAT, 0.5f, 0.2f, F_TMP, nullptr);
  colsum_abs_kernel<<<NT/256,256,0,stream>>>(F_TMP, NT, NT, CS);
  combine_kernel<<<NB,256,0,stream>>>(F_TMP, CS, NT, NT, f_agg_f_w, 2, 0, 0, F_TMP2);
  rownorm_kernel<<<NT,FEAT,0,stream>>>(F_FPS, FEAT, fpo_w, B_Y);
  simgemm_kernel<<<dim3(NT/64, NT/64),128,0,stream>>>(B_Y, NT, 2*FEAT, 0.5f, 0.2f, F_TMP, nullptr);
  colsum_abs_kernel<<<NT/256,256,0,stream>>>(F_TMP, NT, NT, CS);
  combine_kernel<<<NB,256,0,stream>>>(F_TMP, CS, NT, NT, f_agg_f_w, 2, 1, 1, F_TMP2);
  colsum_abs_kernel<<<NT/256,256,0,stream>>>(F_TMP2, NT, NT, CS);
  combine_kernel<<<NB,256,0,stream>>>(F_TMP2, CS, NT, NT, f_agg_w, 4, 2, 1, F_ACC);

  // ---- channel 3: sim_tt = att([gg(topo_hid_a), gg(topo_hid_s)], f_agg_f_w) ----
  rownorm_kernel<<<NT,COM,0,stream>>>(F_THA, COM, fgt_w_a, B_Y);
  simgemm_kernel<<<dim3(NT/64, NT/64),128,0,stream>>>(B_Y, NT, 2*COM, 0.5f, 0.1f, F_TMP, nullptr);
  colsum_abs_kernel<<<NT/256,256,0,stream>>>(F_TMP, NT, NT, CS);
  combine_kernel<<<NB,256,0,stream>>>(F_TMP, CS, NT, NT, f_agg_f_w, 2, 0, 0, F_TMP2);
  rownorm_kernel<<<NT,COM,0,stream>>>(F_THS, COM, fgt_w_s, B_Y);
  simgemm_kernel<<<dim3(NT/64, NT/64),128,0,stream>>>(B_Y, NT, 2*COM, 0.5f, 0.1f, F_TMP, nullptr);
  colsum_abs_kernel<<<NT/256,256,0,stream>>>(F_TMP, NT, NT, CS);
  combine_kernel<<<NB,256,0,stream>>>(F_TMP, CS, NT, NT, f_agg_f_w, 2, 1, 1, F_TMP2);
  colsum_abs_kernel<<<NT/256,256,0,stream>>>(F_TMP2, NT, NT, CS);
  combine_kernel<<<NB,256,0,stream>>>(F_TMP2, CS, NT, NT, f_agg_w, 4, 3, 1, F_ACC);

  // ---- finalize adjacency: symmetrize + column L1 normalize ----
  symmetrize_kernel<<<dim3(NT/256, NT),256,0,stream>>>(F_ACC, NT, F_TMP);
  colsum_abs_kernel<<<NT/256,256,0,stream>>>(F_TMP, NT, NT, CS);
  normadj_kernel<<<NB,256,0,stream>>>(F_TMP, CS, NT, adj_out, B_SIMA);

  // ---- GCN (eval mode) ----
  gemm_kernel<<<dim3(1, NT/64),128,0,stream>>>(B_FT, B_W1T, NT, EMB, FEAT, nullptr, 0, F_H0);
  transpose_bf16_kernel<<<grid1((size_t)NT*EMB),256,0,stream>>>(F_H0, NT, EMB, B_H0T);
  gemm_kernel<<<dim3(1, NT/64),128,0,stream>>>(B_SIMA, B_H0T, NT, EMB, NT, gcn_b1, 1, F_H1);
  small_gemm_kernel<<<grid1((size_t)NT*NCLS),256,0,stream>>>(F_H1, gcn_W2, NT, EMB, NCLS, F_X2);
  final_logits_kernel<<<NT,128,0,stream>>>(adj_out, F_X2, gcn_b2, NT, logits);
}
